// EnhancedGNN_9766755631324
// MI455X (gfx1250) — matrix operand order repaired and checked by reading, not compiled, not run on hardware
//
#include <hip/hip_runtime.h>
#include <math.h>

#define N_NODES 50000
#define N_EDGES 800000
#define N_GRAPH 512
#define F_IN    64
#define D_GLOB  32
#define H_DIM   128
#define LAYERS  3

#define CDIV(a,b) (((a)+(b)-1)/(b))

typedef __attribute__((ext_vector_type(16))) __bf16 v16bf;
typedef __attribute__((ext_vector_type(8)))  float  v8f;

union BF16Frag { unsigned int u[8]; uint4 q[2]; v16bf v; };

__device__ __forceinline__ unsigned short f2bf(float f) {
    unsigned int u = __float_as_uint(f);
    u += 0x7FFFu + ((u >> 16) & 1u);            // round-to-nearest-even
    return (unsigned short)(u >> 16);
}

__device__ __forceinline__ unsigned int pack2bf(float a, float b) {
    return (unsigned int)f2bf(a) | ((unsigned int)f2bf(b) << 16);
}

__device__ __forceinline__ float lrelu(float x) { return x > 0.f ? x : 0.2f * x; }

// float atomic max via sign-aware integer ordering (init must be -inf)
__device__ __forceinline__ void atomicMaxF(float* a, float v) {
    if (v >= 0.f) atomicMax((int*)a, __float_as_int(v));
    else          atomicMin((unsigned int*)a, (unsigned int)__float_as_int(v));
}

// ---------------------------------------------------------------------------
// Prep kernels
// ---------------------------------------------------------------------------

// eval-BN:  y = x*scale + shift,  scale = g*rsqrt(v+eps), shift = b - m*scale
__global__ void bn_prep_kernel(const float* __restrict__ g, const float* __restrict__ b,
                               const float* __restrict__ m, const float* __restrict__ v,
                               float* __restrict__ scale, float* __restrict__ shift, int n) {
    int i = blockIdx.x * blockDim.x + threadIdx.x;
    if (i < n) {
        float s = g[i] * rsqrtf(v[i] + 1e-5f);
        scale[i] = s;
        shift[i] = b[i] - m[i] * s;
    }
}

// vectorized x4: n must be a multiple of 4 (true for all uses here)
__global__ void f32_to_bf16_v4_kernel(const float* __restrict__ in,
                                      unsigned int* __restrict__ out, long long n4) {
    long long i = (long long)blockIdx.x * blockDim.x + threadIdx.x;
    if (i >= n4) return;
    float4 v = ((const float4*)in)[i];
    uint2 p;
    p.x = pack2bf(v.x, v.y);
    p.y = pack2bf(v.z, v.w);
    ((uint2*)out)[i] = p;
}

// Pack W (fp32 row-major [K][128]) into per-lane WMMA B-fragments (bf16 pairs).
// Layout: frag[((nt*kSteps + kt)*32 + lane)*8 + p], pair = (W[k][n], W[k+1][n]),
// k = kt*32 + 8*(lane>>4) + (p<4 ? 2p : 8+2p), n = nt*16 + (lane&15).
__global__ void pack_bfrag_kernel(const float* __restrict__ W,
                                  unsigned int* __restrict__ frag, int K) {
    int kSteps = K >> 5;
    int total = 8 * kSteps * 32 * 8;
    int idx = blockIdx.x * blockDim.x + threadIdx.x;
    if (idx >= total) return;
    int p    = idx & 7;
    int lane = (idx >> 3) & 31;
    int kt   = (idx >> 8) % kSteps;
    int nt   = (idx >> 8) / kSteps;
    int k = kt * 32 + 8 * (lane >> 4) + (p < 4 ? 2 * p : 8 + 2 * p);
    int n = nt * 16 + (lane & 15);
    frag[idx] = pack2bf(W[(size_t)k * H_DIM + n], W[(size_t)(k + 1) * H_DIM + n]);
}

// ---------------------------------------------------------------------------
// WMMA GEMM:  out[M x 128] = A_bf16[M x K] @ W  (+bias, *bn, relu, bf16 copy)
//
// Block = 128 threads = 4 waves; block owns 16 M-rows; wave w owns N-tiles
// {2w, 2w+1}.  Per the ISA 16-bit A layout, lane (m=lane&15, half=lane>>4)
// holds K in [8*half, 8*half+8) and [16+8*half, ...+8): two CONTIGUOUS 16-byte
// chunks of row m, so the A fragment is built with two global_load_b128 — no
// LDS, no barriers in the k-loop.  One A fragment feeds two independent WMMA
// accumulator chains.
// ---------------------------------------------------------------------------
__global__ __launch_bounds__(128)
void gemm_bf16_wmma_kernel(const unsigned short* __restrict__ A,
                           const unsigned int*  __restrict__ Bfrag,
                           float* __restrict__ out,
                           unsigned short* __restrict__ out_bf,   // optional
                           int K,
                           const float* __restrict__ bias,        // optional
                           const float* __restrict__ bn_scale,    // optional
                           const float* __restrict__ bn_shift,    // optional
                           int relu) {
    const int tid  = threadIdx.x;
    const int wave = tid >> 5;                  // 0..3
    const int lane = tid & 31;
    const int m0   = blockIdx.x * 16;
    const int nt0  = wave * 2;                  // first of two n-tiles
    const int kSteps = K >> 5;

    const int mrow = lane & 15;
    const int half = lane >> 4;

    // A: row-major bf16; per-lane fragment base (16B aligned: K multiple of 16)
    const unsigned short* abase = A + (size_t)(m0 + mrow) * K + 8 * half;
    const unsigned int* bp0 = Bfrag + ((size_t)(nt0)     * kSteps * 32 + lane) * 8;
    const unsigned int* bp1 = Bfrag + ((size_t)(nt0 + 1) * kSteps * 32 + lane) * 8;

    v8f acc0 = {}, acc1 = {};

    for (int kt = 0; kt < kSteps; ++kt) {
        BF16Frag af, b0, b1;
        const uint4* ap = (const uint4*)(abase + kt * 32);
        af.q[0] = ap[0];                        // K = 8*half .. +8
        af.q[1] = ap[2];                        // K = 16+8*half .. +8
        const uint4* q0 = (const uint4*)(bp0 + (size_t)kt * 32 * 8);
        const uint4* q1 = (const uint4*)(bp1 + (size_t)kt * 32 * 8);
        b0.q[0] = q0[0]; b0.q[1] = q0[1];
        b1.q[0] = q1[0]; b1.q[1] = q1[1];

        acc0 = __builtin_amdgcn_wmma_f32_16x16x32_bf16(
                   false, af.v, false, b0.v, (short)0, acc0, false, false);
        acc1 = __builtin_amdgcn_wmma_f32_16x16x32_bf16(
                   false, af.v, false, b1.v, (short)0, acc1, false, false);
    }

    // C/D layout: lane n = lane&15; M base = (lane>>4)*8; VGPR r -> M = base+r
    const int n  = lane & 15;
    const int mb = half * 8;
    #pragma unroll
    for (int j = 0; j < 2; ++j) {
        const int cn = (nt0 + j) * 16 + n;
        const float bia = bias     ? bias[cn]     : 0.f;
        const float sc  = bn_scale ? bn_scale[cn] : 1.f;
        const float sh  = bn_shift ? bn_shift[cn] : 0.f;
        const v8f acc = j ? acc1 : acc0;
        #pragma unroll
        for (int r = 0; r < 8; ++r) {
            float v = (acc[r] + bia) * sc + sh;
            if (relu) v = fmaxf(v, 0.f);
            size_t o = (size_t)(m0 + mb + r) * H_DIM + cn;
            out[o] = v;
            if (out_bf) out_bf[o] = f2bf(v);
        }
    }
}

// ---------------------------------------------------------------------------
// GAT attention kernels
// ---------------------------------------------------------------------------
__global__ void attn_coef_kernel(const float* __restrict__ hh,
                                 const float* __restrict__ wsrc,
                                 const float* __restrict__ wdst,
                                 float* __restrict__ a_src,
                                 float* __restrict__ a_dst, int n) {
    int idx = blockIdx.x * blockDim.x + threadIdx.x;   // (node, head)
    if (idx >= n * 4) return;
    int node = idx >> 2, hd = idx & 3;
    const float* v  = hh + (size_t)node * H_DIM + hd * 32;
    const float* w1 = wsrc + hd * 32;
    const float* w2 = wdst + hd * 32;
    float s1 = 0.f, s2 = 0.f;
    #pragma unroll
    for (int c = 0; c < 32; ++c) { float x = v[c]; s1 += x * w1[c]; s2 += x * w2[c]; }
    a_src[idx] = s1;
    a_dst[idx] = s2;
}

// one thread per 4 floats: agg zeros (N*128), and for idx<N emax/den float4 init
__global__ void layer_init_v4_kernel(float* __restrict__ agg, float* __restrict__ emax,
                                     float* __restrict__ den) {
    long long idx = (long long)blockIdx.x * blockDim.x + threadIdx.x;
    if (idx >= (long long)N_NODES * 32) return;
    ((float4*)agg)[idx] = make_float4(0.f, 0.f, 0.f, 0.f);
    if (idx < N_NODES) {
        ((float4*)emax)[idx] = make_float4(-INFINITY, -INFINITY, -INFINITY, -INFINITY);
        ((float4*)den)[idx]  = make_float4(0.f, 0.f, 0.f, 0.f);
    }
}

// edges 0..E-1 are real, E..E+N-1 are self loops
__global__ void edge_max_kernel(const int* __restrict__ src, const int* __restrict__ dst,
                                const float* __restrict__ a_src, const float* __restrict__ a_dst,
                                float* __restrict__ emax) {
    int e = blockIdx.x * blockDim.x + threadIdx.x;
    if (e >= N_EDGES + N_NODES) return;
    int s, d;
    if (e < N_EDGES) { s = src[e]; d = dst[e]; } else { s = d = e - N_EDGES; }
    const float4 as = *(const float4*)(a_src + (size_t)s * 4);
    const float4 ad = *(const float4*)(a_dst + (size_t)d * 4);
    atomicMaxF(&emax[d * 4 + 0], lrelu(as.x + ad.x));
    atomicMaxF(&emax[d * 4 + 1], lrelu(as.y + ad.y));
    atomicMaxF(&emax[d * 4 + 2], lrelu(as.z + ad.z));
    atomicMaxF(&emax[d * 4 + 3], lrelu(as.w + ad.w));
}

__global__ void edge_sum_kernel(const int* __restrict__ src, const int* __restrict__ dst,
                                const float* __restrict__ a_src, const float* __restrict__ a_dst,
                                const float* __restrict__ emax, float* __restrict__ den) {
    int e = blockIdx.x * blockDim.x + threadIdx.x;
    if (e >= N_EDGES + N_NODES) return;
    int s, d;
    if (e < N_EDGES) { s = src[e]; d = dst[e]; } else { s = d = e - N_EDGES; }
    const float4 as = *(const float4*)(a_src + (size_t)s * 4);
    const float4 ad = *(const float4*)(a_dst + (size_t)d * 4);
    const float4 em = *(const float4*)(emax + (size_t)d * 4);
    atomicAdd(&den[d * 4 + 0], __expf(lrelu(as.x + ad.x) - em.x));
    atomicAdd(&den[d * 4 + 1], __expf(lrelu(as.y + ad.y) - em.y));
    atomicAdd(&den[d * 4 + 2], __expf(lrelu(as.z + ad.z) - em.z));
    atomicAdd(&den[d * 4 + 3], __expf(lrelu(as.w + ad.w) - em.w));
}

// 32 threads per edge; each thread: 4 consecutive columns (single head),
// one alpha computation, one float4 gather, 4 f32 atomic adds.
__global__ void edge_aggr_v4_kernel(const int* __restrict__ src, const int* __restrict__ dst,
                                    const float* __restrict__ a_src, const float* __restrict__ a_dst,
                                    const float* __restrict__ emax, const float* __restrict__ den,
                                    const float* __restrict__ hh, float* __restrict__ agg) {
    long long gid = (long long)blockIdx.x * blockDim.x + threadIdx.x;
    if (gid >= (long long)(N_EDGES + N_NODES) * 32) return;
    int t4 = (int)(gid & 31);           // 4-column group 0..31
    int e  = (int)(gid >> 5);
    int s, d;
    if (e < N_EDGES) { s = src[e]; d = dst[e]; } else { s = d = e - N_EDGES; }
    int col0 = t4 * 4;
    int hd   = t4 >> 3;                 // 4-col group lies inside one 32-col head
    float v = lrelu(a_src[(size_t)s * 4 + hd] + a_dst[(size_t)d * 4 + hd]);
    float alpha = __expf(v - emax[(size_t)d * 4 + hd]) / (den[(size_t)d * 4 + hd] + 1e-16f);
    const float4 hv = *(const float4*)(hh + (size_t)s * H_DIM + col0);
    float* ap = agg + (size_t)d * H_DIM + col0;
    atomicAdd(ap + 0, hv.x * alpha);
    atomicAdd(ap + 1, hv.y * alpha);
    atomicAdd(ap + 2, hv.z * alpha);
    atomicAdd(ap + 3, hv.w * alpha);
}

// out = relu(bn(agg + bias)); h = out + h; also emit bf16 for next GEMM (x4)
__global__ void gat_post_v4_kernel(float* __restrict__ h, unsigned int* __restrict__ h_bf,
                                   const float* __restrict__ agg, const float* __restrict__ bias,
                                   const float* __restrict__ scale, const float* __restrict__ shift) {
    long long idx = (long long)blockIdx.x * blockDim.x + threadIdx.x;
    if (idx >= (long long)N_NODES * 32) return;
    int col0 = (int)(idx & 31) * 4;
    float4 a = ((const float4*)agg)[idx];
    float4 hv = ((const float4*)h)[idx];
    const float4 bi = *(const float4*)(bias + col0);
    const float4 sc = *(const float4*)(scale + col0);
    const float4 sh = *(const float4*)(shift + col0);
    float4 o;
    o.x = fmaxf((a.x + bi.x) * sc.x + sh.x, 0.f) + hv.x;
    o.y = fmaxf((a.y + bi.y) * sc.y + sh.y, 0.f) + hv.y;
    o.z = fmaxf((a.z + bi.z) * sc.z + sh.z, 0.f) + hv.z;
    o.w = fmaxf((a.w + bi.w) * sc.w + sh.w, 0.f) + hv.w;
    ((float4*)h)[idx] = o;
    uint2 p; p.x = pack2bf(o.x, o.y); p.y = pack2bf(o.z, o.w);
    ((uint2*)h_bf)[idx] = p;
}

// ---------------------------------------------------------------------------
// Pooling + readout
// ---------------------------------------------------------------------------
__global__ void pool_init_kernel(float* __restrict__ psum, float* __restrict__ pmax,
                                 float* __restrict__ cnt) {
    int idx = blockIdx.x * blockDim.x + threadIdx.x;
    if (idx >= N_GRAPH * H_DIM) return;
    psum[idx] = 0.f;
    pmax[idx] = -INFINITY;
    if (idx < N_GRAPH) cnt[idx] = 0.f;
}

// 32 threads per node, float4 per thread
__global__ void pool_accum_v4_kernel(const float* __restrict__ h, const int* __restrict__ batch,
                                     float* __restrict__ psum, float* __restrict__ pmax,
                                     float* __restrict__ cnt) {
    long long idx = (long long)blockIdx.x * blockDim.x + threadIdx.x;
    if (idx >= (long long)N_NODES * 32) return;
    int node = (int)(idx >> 5), t4 = (int)(idx & 31);
    int col0 = t4 * 4;
    int g = batch[node];
    float4 v = ((const float4*)h)[idx];
    float* ps = psum + (size_t)g * H_DIM + col0;
    float* pm = pmax + (size_t)g * H_DIM + col0;
    atomicAdd(ps + 0, v.x);  atomicMaxF(pm + 0, v.x);
    atomicAdd(ps + 1, v.y);  atomicMaxF(pm + 1, v.y);
    atomicAdd(ps + 2, v.z);  atomicMaxF(pm + 2, v.z);
    atomicAdd(ps + 3, v.w);  atomicMaxF(pm + 3, v.w);
    if (t4 == 0) atomicAdd(&cnt[g], 1.0f);
}

// z_bf16[G][384] = bf16(concat[mean, max(fixed), gfeat])
__global__ void build_z_kernel(const float* __restrict__ psum, const float* __restrict__ pmax,
                               const float* __restrict__ cnt, const float* __restrict__ gfeat,
                               unsigned short* __restrict__ z_bf) {
    int idx = blockIdx.x * blockDim.x + threadIdx.x;
    if (idx >= N_GRAPH * 384) return;
    int g = idx / 384, c = idx % 384;
    float v;
    if (c < 128)      v = psum[(size_t)g * H_DIM + c] / fmaxf(cnt[g], 1.f);
    else if (c < 256) { v = pmax[(size_t)g * H_DIM + (c - 128)]; if (!isfinite(v)) v = 0.f; }
    else              v = gfeat[(size_t)g * H_DIM + (c - 256)];
    z_bf[idx] = f2bf(v);
}

// out[g] = dot(z1[g], lin2_w) + lin2_b   (NC == 1)
__global__ __launch_bounds__(128)
void final_dot_kernel(const float* __restrict__ z1, const float* __restrict__ w,
                      const float* __restrict__ b, float* __restrict__ out) {
    __shared__ float red[128];
    int g = blockIdx.x, t = threadIdx.x;
    red[t] = z1[(size_t)g * H_DIM + t] * w[t];
    __syncthreads();
    for (int s = 64; s > 0; s >>= 1) { if (t < s) red[t] += red[t + s]; __syncthreads(); }
    if (t == 0) out[g] = red[0] + b[0];
}

// ---------------------------------------------------------------------------
// Host launcher
// ---------------------------------------------------------------------------
extern "C" void kernel_launch(void* const* d_in, const int* in_sizes, int n_in,
                              void* d_out, int out_size, void* d_ws, size_t ws_size,
                              hipStream_t stream) {
    const float* x      = (const float*)d_in[0];
    const float* gfeat0 = (const float*)d_in[2];
    const int*   eidx   = (const int*)d_in[3];
    const int*   batch  = (const int*)d_in[4];
    const float* node_w = (const float*)d_in[5];
    const float* node_b = (const float*)d_in[6];
    const float* nbn_g  = (const float*)d_in[7];
    const float* nbn_b  = (const float*)d_in[8];
    const float* nbn_m  = (const float*)d_in[9];
    const float* nbn_v  = (const float*)d_in[10];
    const float* glob_w = (const float*)d_in[13];
    const float* glob_b = (const float*)d_in[14];
    const float* gbn_g  = (const float*)d_in[15];
    const float* gbn_b  = (const float*)d_in[16];
    const float* gbn_m  = (const float*)d_in[17];
    const float* gbn_v  = (const float*)d_in[18];
    const float* gat_lin  = (const float*)d_in[19];
    const float* gat_asrc = (const float*)d_in[20];
    const float* gat_adst = (const float*)d_in[21];
    const float* gat_bias = (const float*)d_in[22];
    const float* bn_g   = (const float*)d_in[23];
    const float* bn_b   = (const float*)d_in[24];
    const float* bn_m   = (const float*)d_in[25];
    const float* bn_v   = (const float*)d_in[26];
    const float* lin1_w = (const float*)d_in[27];
    const float* lin1_b = (const float*)d_in[28];
    const float* lin2_w = (const float*)d_in[29];
    const float* lin2_b = (const float*)d_in[30];

    const int* src = eidx;
    const int* dst = eidx + N_EDGES;

    // ---- workspace carve (256B aligned) ----
    char* wsp = (char*)d_ws;
    auto carve = [&](size_t bytes) -> void* {
        void* p = (void*)wsp;
        wsp += (bytes + 255) & ~(size_t)255;
        return p;
    };
    float*          h      = (float*)carve((size_t)N_NODES * H_DIM * 4);
    unsigned short* h_bf   = (unsigned short*)carve((size_t)N_NODES * H_DIM * 2);
    float*          hh     = (float*)carve((size_t)N_NODES * H_DIM * 4);
    float*          agg    = (float*)carve((size_t)N_NODES * H_DIM * 4);
    float*          a_src  = (float*)carve((size_t)N_NODES * 4 * 4);
    float*          a_dst  = (float*)carve((size_t)N_NODES * 4 * 4);
    float*          emax   = (float*)carve((size_t)N_NODES * 4 * 4);
    float*          den    = (float*)carve((size_t)N_NODES * 4 * 4);
    unsigned short* x_bf   = (unsigned short*)carve((size_t)N_NODES * F_IN * 2);
    unsigned short* gf_bf  = (unsigned short*)carve((size_t)N_GRAPH * D_GLOB * 2);
    float*          gfeat  = (float*)carve((size_t)N_GRAPH * H_DIM * 4);
    float*          psum   = (float*)carve((size_t)N_GRAPH * H_DIM * 4);
    float*          pmax   = (float*)carve((size_t)N_GRAPH * H_DIM * 4);
    float*          cnt    = (float*)carve((size_t)N_GRAPH * 4);
    unsigned short* z_bf   = (unsigned short*)carve((size_t)N_GRAPH * 384 * 2);
    float*          z1     = (float*)carve((size_t)N_GRAPH * H_DIM * 4);
    unsigned int*   fr_node = (unsigned int*)carve(8 * (F_IN / 32) * 32 * 8 * 4);
    unsigned int*   fr_gat  = (unsigned int*)carve((size_t)LAYERS * 8 * (H_DIM / 32) * 32 * 8 * 4);
    unsigned int*   fr_glob = (unsigned int*)carve(8 * (D_GLOB / 32) * 32 * 8 * 4);
    unsigned int*   fr_lin1 = (unsigned int*)carve(8 * (384 / 32) * 32 * 8 * 4);
    float* nbn_s = (float*)carve(H_DIM * 4);  float* nbn_t = (float*)carve(H_DIM * 4);
    float* gbn_s = (float*)carve(H_DIM * 4);  float* gbn_t = (float*)carve(H_DIM * 4);
    float* lbn_s = (float*)carve((size_t)LAYERS * H_DIM * 4);
    float* lbn_t = (float*)carve((size_t)LAYERS * H_DIM * 4);

    // ---- prep: BN folds + weight fragment packing + bf16 converts ----
    bn_prep_kernel<<<1, 128, 0, stream>>>(nbn_g, nbn_b, nbn_m, nbn_v, nbn_s, nbn_t, H_DIM);
    bn_prep_kernel<<<1, 128, 0, stream>>>(gbn_g, gbn_b, gbn_m, gbn_v, gbn_s, gbn_t, H_DIM);
    for (int i = 0; i < LAYERS; ++i)
        bn_prep_kernel<<<1, 128, 0, stream>>>(bn_g + i * H_DIM, bn_b + i * H_DIM,
                                              bn_m + i * H_DIM, bn_v + i * H_DIM,
                                              lbn_s + i * H_DIM, lbn_t + i * H_DIM, H_DIM);

    pack_bfrag_kernel<<<CDIV(8 * (F_IN / 32) * 256, 256), 256, 0, stream>>>(node_w, fr_node, F_IN);
    for (int i = 0; i < LAYERS; ++i)
        pack_bfrag_kernel<<<CDIV(8 * (H_DIM / 32) * 256, 256), 256, 0, stream>>>(
            gat_lin + (size_t)i * H_DIM * H_DIM, fr_gat + (size_t)i * 8 * (H_DIM / 32) * 256, H_DIM);
    pack_bfrag_kernel<<<CDIV(8 * (D_GLOB / 32) * 256, 256), 256, 0, stream>>>(glob_w, fr_glob, D_GLOB);
    pack_bfrag_kernel<<<CDIV(8 * (384 / 32) * 256, 256), 256, 0, stream>>>(lin1_w, fr_lin1, 384);

    f32_to_bf16_v4_kernel<<<CDIV((long long)N_NODES * F_IN / 4, 256), 256, 0, stream>>>(
        x, (unsigned int*)x_bf, (long long)N_NODES * F_IN / 4);
    f32_to_bf16_v4_kernel<<<CDIV(N_GRAPH * D_GLOB / 4, 256), 256, 0, stream>>>(
        gfeat0, (unsigned int*)gf_bf, (long long)N_GRAPH * D_GLOB / 4);

    // ---- node encoder: h = relu(bn(x @ node_w + b)), also emits h_bf16 ----
    gemm_bf16_wmma_kernel<<<N_NODES / 16, 128, 0, stream>>>(
        x_bf, fr_node, h, h_bf, F_IN, node_b, nbn_s, nbn_t, 1);

    // ---- GAT layers ----
    const long long nodes32 = (long long)N_NODES * 32;
    const long long edges32 = (long long)(N_EDGES + N_NODES) * 32;
    for (int i = 0; i < LAYERS; ++i) {
        gemm_bf16_wmma_kernel<<<N_NODES / 16, 128, 0, stream>>>(
            h_bf, fr_gat + (size_t)i * 8 * (H_DIM / 32) * 256, hh, nullptr,
            H_DIM, nullptr, nullptr, nullptr, 0);
        attn_coef_kernel<<<CDIV(N_NODES * 4, 256), 256, 0, stream>>>(
            hh, gat_asrc + i * 128, gat_adst + i * 128, a_src, a_dst, N_NODES);
        layer_init_v4_kernel<<<CDIV(nodes32, 256), 256, 0, stream>>>(agg, emax, den);
        edge_max_kernel<<<CDIV(N_EDGES + N_NODES, 256), 256, 0, stream>>>(src, dst, a_src, a_dst, emax);
        edge_sum_kernel<<<CDIV(N_EDGES + N_NODES, 256), 256, 0, stream>>>(src, dst, a_src, a_dst, emax, den);
        edge_aggr_v4_kernel<<<CDIV(edges32, 256), 256, 0, stream>>>(src, dst, a_src, a_dst, emax, den, hh, agg);
        gat_post_v4_kernel<<<CDIV(nodes32, 256), 256, 0, stream>>>(
            h, (unsigned int*)h_bf, agg, gat_bias + i * H_DIM, lbn_s + i * H_DIM, lbn_t + i * H_DIM);
    }

    // ---- pooling ----
    pool_init_kernel<<<CDIV(N_GRAPH * H_DIM, 256), 256, 0, stream>>>(psum, pmax, cnt);
    pool_accum_v4_kernel<<<CDIV(nodes32, 256), 256, 0, stream>>>(h, batch, psum, pmax, cnt);

    // ---- global branch: gfeat = relu(bn(gf @ glob_w + b)) ----
    gemm_bf16_wmma_kernel<<<N_GRAPH / 16, 128, 0, stream>>>(
        gf_bf, fr_glob, gfeat, nullptr, D_GLOB, glob_b, gbn_s, gbn_t, 1);

    // ---- readout: z = concat -> lin1 -> relu -> lin2 ----
    build_z_kernel<<<CDIV(N_GRAPH * 384, 256), 256, 0, stream>>>(psum, pmax, cnt, gfeat, z_bf);
    gemm_bf16_wmma_kernel<<<N_GRAPH / 16, 128, 0, stream>>>(
        z_bf, fr_lin1, z1, nullptr, 384, lin1_b, nullptr, nullptr, 1);
    final_dot_kernel<<<N_GRAPH, 128, 0, stream>>>(z1, lin2_w, lin2_b, (float*)d_out);

    (void)in_sizes; (void)n_in; (void)out_size; (void)ws_size;
}
